// ImMatchNet_71390946394719
// MI455X (gfx1250) — matrix operand order repaired and checked by reading, not compiled, not run on hardware
//
#include <hip/hip_runtime.h>

// ---------------------------------------------------------------------------
// ImMatchNet pipeline for MI455X (gfx1250, wave32, WMMA)
//   B=8, C=1024, H=W=32, K=2
// Stages:
//   1) inv-norm over C per (b,hw)
//   2) normalize + transpose + fp32->bf16  => AT/BT laid out (spatial, channel)
//   3) batched GEMM 1024x1024x1024 with v_wmma_f32_16x16x32_bf16
//   4) 2^4 maxpool -> (16,16,16,16)
//   5) mutual matching (two max reductions + elementwise)
//   6) 3-layer 4D conv net as implicit GEMM on WMMA (LDS-staged halos),
//      on x and on (ij)<->(pq) transposed x, summed
//   7) mutual matching -> d_out
// ---------------------------------------------------------------------------

typedef __attribute__((ext_vector_type(16))) __bf16 v16bf;
typedef __attribute__((ext_vector_type(8)))  float  v8f;

__device__ __forceinline__ unsigned short f2bf(float f) {
    unsigned int u = __float_as_uint(f);
    unsigned int r = 0x7FFFu + ((u >> 16) & 1u);   // round-to-nearest-even
    return (unsigned short)((u + r) >> 16);
}

// ---------------- 1) inverse norm over channels ----------------------------
__global__ void invnorm_kernel(const float* __restrict__ f, float* __restrict__ inv) {
    int b  = blockIdx.y;
    int hw = blockIdx.x * 256 + threadIdx.x;
    const float* p = f + ((size_t)b * 1024u) * 1024u + hw;
    float s = 0.f;
    #pragma unroll 4
    for (int c = 0; c < 1024; ++c) s = fmaf(p[(size_t)c * 1024u], p[(size_t)c * 1024u], s);
    inv[b * 1024 + hw] = rsqrtf(s + 1e-6f);
}

// ------------- 2) normalize + transpose (c,hw)->(hw,c), fp32->bf16 ---------
__global__ void norm_transpose_kernel(const float* __restrict__ f,
                                      const float* __restrict__ inv,
                                      unsigned short* __restrict__ out) {
    __shared__ unsigned short tile[32][33];
    int b = blockIdx.z, c0 = blockIdx.y * 32, hw0 = blockIdx.x * 32;
    int tx = threadIdx.x, ty = threadIdx.y;
    float s = inv[b * 1024 + hw0 + tx];
    #pragma unroll
    for (int r = 0; r < 4; ++r) {
        int c = c0 + ty + r * 8;
        float v = f[((size_t)(b * 1024 + c)) * 1024u + hw0 + tx] * s;
        tile[tx][ty + r * 8] = f2bf(v);
    }
    __syncthreads();
    #pragma unroll
    for (int r = 0; r < 4; ++r) {
        int hw = hw0 + ty + r * 8;
        out[((size_t)(b * 1024 + hw)) * 1024u + c0 + tx] = tile[ty + r * 8][tx];
    }
}

// ---------------- 3) batched bf16 WMMA GEMM --------------------------------
// AT, BT: (b, 1024 spatial rows, 1024 channels) bf16, row-major in channel.
// corr[b][m][n] = sum_k AT[b][m][k] * BT[b][n][k]
// Block: 128 threads = 4 waves (2x2), wave tile 64x64 = 4x4 WMMA accumulators.
__global__ void __launch_bounds__(128)
corr_gemm_kernel(const unsigned short* __restrict__ AT,
                 const unsigned short* __restrict__ BT,
                 float* __restrict__ corr) {
    union Frag { uint4 u[2]; v16bf v; };

    const int lane  = threadIdx.x & 31;
    const int wave  = threadIdx.x >> 5;
    const int half  = lane >> 4;        // 0: lanes 0-15, 1: lanes 16-31
    const int lrow  = lane & 15;
    const int b     = blockIdx.z;
    const int M0    = blockIdx.x * 128 + (wave >> 1) * 64;
    const int N0    = blockIdx.y * 128 + (wave & 1) * 64;

    const unsigned short* Ab = AT + ((size_t)b << 20);
    const unsigned short* Bb = BT + ((size_t)b << 20);

    v8f acc[4][4];
    #pragma unroll
    for (int i = 0; i < 4; ++i)
        #pragma unroll
        for (int j = 0; j < 4; ++j)
            acc[i][j] = (v8f){0.f, 0.f, 0.f, 0.f, 0.f, 0.f, 0.f, 0.f};

    for (int k = 0; k < 1024; k += 32) {
        Frag a[4], bb[4];
        #pragma unroll
        for (int i = 0; i < 4; ++i) {
            const unsigned short* p =
                Ab + (size_t)(M0 + i * 16 + lrow) * 1024u + k + half * 8;
            a[i].u[0] = *reinterpret_cast<const uint4*>(p);
            a[i].u[1] = *reinterpret_cast<const uint4*>(p + 16);
        }
        #pragma unroll
        for (int j = 0; j < 4; ++j) {
            const unsigned short* p =
                Bb + (size_t)(N0 + j * 16 + lrow) * 1024u + k + half * 8;
            bb[j].u[0] = *reinterpret_cast<const uint4*>(p);
            bb[j].u[1] = *reinterpret_cast<const uint4*>(p + 16);
        }
        #pragma unroll
        for (int i = 0; i < 4; ++i)
            #pragma unroll
            for (int j = 0; j < 4; ++j)
                acc[i][j] = __builtin_amdgcn_wmma_f32_16x16x32_bf16(
                    false, a[i].v, false, bb[j].v, (short)0, acc[i][j],
                    false, false);
    }

    float* Cb = corr + ((size_t)b << 20);
    #pragma unroll
    for (int i = 0; i < 4; ++i)
        #pragma unroll
        for (int j = 0; j < 4; ++j) {
            int col = N0 + j * 16 + lrow;
            #pragma unroll
            for (int r = 0; r < 8; ++r) {
                int row = M0 + i * 16 + half * 8 + r;
                Cb[(size_t)row * 1024u + col] = acc[i][j][r];
            }
        }
}

// ---------------- 4) 2x2x2x2 max pool --------------------------------------
__global__ void maxpool4d_kernel(const float* __restrict__ corr,
                                 float* __restrict__ pooled) {
    int idx = blockIdx.x * 256 + threadIdx.x;       // 524288 total
    int q = idx & 15, p = (idx >> 4) & 15, j = (idx >> 8) & 15,
        i = (idx >> 12) & 15, b = idx >> 16;
    const float* cb = corr + ((size_t)b << 20);
    float m = -3.4e38f;
    #pragma unroll
    for (int di = 0; di < 2; ++di)
        #pragma unroll
        for (int dj = 0; dj < 2; ++dj) {
            int hw = (2 * i + di) * 32 + (2 * j + dj);
            const float* row = cb + (size_t)hw * 1024u;
            #pragma unroll
            for (int dp = 0; dp < 2; ++dp)
                #pragma unroll
                for (int dq = 0; dq < 2; ++dq) {
                    float v = row[(2 * p + dp) * 32 + (2 * q + dq)];
                    m = fmaxf(m, v);
                }
        }
    pooled[idx] = m;
}

// -------- 5/7) mutual matching: reductions + elementwise -------------------
__global__ void reduce_last_kernel(const float* __restrict__ x,
                                   float* __restrict__ maxA) {
    __shared__ float sm[256];
    int row = blockIdx.x;                            // 2048 rows
    int t = threadIdx.x;
    sm[t] = x[(size_t)row * 256u + t];
    __syncthreads();
    for (int off = 128; off > 0; off >>= 1) {
        if (t < off) sm[t] = fmaxf(sm[t], sm[t + off]);
        __syncthreads();
    }
    if (t == 0) maxA[row] = sm[0];
}

__global__ void reduce_mid_kernel(const float* __restrict__ x,
                                  float* __restrict__ maxB) {
    int b = blockIdx.x, t = threadIdx.x;
    const float* xb = x + ((size_t)b << 16);
    float m = -3.4e38f;
    for (int s = 0; s < 256; ++s) m = fmaxf(m, xb[s * 256 + t]);
    maxB[b * 256 + t] = m;
}

__global__ void mm_apply_kernel(const float* __restrict__ x,
                                const float* __restrict__ maxA,
                                const float* __restrict__ maxB,
                                float* __restrict__ out) {
    int idx = blockIdx.x * 256 + threadIdx.x;
    int b = idx >> 16, s = (idx >> 8) & 255, t = idx & 255;
    float c = x[idx];
    float ra = c / (maxA[b * 256 + s] + 1e-5f);
    float rb = c / (maxB[b * 256 + t] + 1e-5f);
    out[idx] = c * ra * rb;
}

// --------- 6) 4D conv (3^4 SAME) + bias + ReLU as implicit-GEMM WMMA -------
// x: (8, CI, 16,16,16,16), w: (CO, CI, 81), y: (8, CO, 16^4)
// Block = one (b, i, j): M = 256 (p,q) outputs, N = co padded to 16,
// K = CI*81 padded to KP (mult of 32). 256 threads = 8 waves; each wave owns
// two 16-row M tiles. Input halos staged in LDS as zero-padded 18x18 bf16
// planes + a dedicated zero plane for K padding; im2col gathers are single
// ds_load_u16 at (pq_base + offtab[k]).
__global__ void __launch_bounds__(256)
conv4d_wmma_kernel(const float* __restrict__ x, const float* __restrict__ w,
                   const float* __restrict__ bias, float* __restrict__ y,
                   int CI, int CO) {
    extern __shared__ __align__(16) unsigned short smem[];
    const int tid = threadIdx.x;
    const int b = blockIdx.z;
    const int ij = blockIdx.x;            // i*16 + j
    const int i = ij >> 4, j = ij & 15;

    const int KC      = CI * 81;
    const int KP      = ((KC + 31) >> 5) << 5;
    const int CI9     = CI * 9;
    const int planesz = (CI9 + 1) * 324;          // + zero plane
    const int wboff   = (planesz + 7) & ~7;       // 16B-aligned halfword offset
    const int otoff   = wboff + 16 * KP;

    // ---- zero all planes (borders + zero plane stay 0) ----
    for (int idx = tid; idx < planesz; idx += 256) smem[idx] = 0;
    __syncthreads();
    // ---- fill plane interiors: plane = ci*9 + di*3 + dj, padded (p+1,q+1) --
    for (int idx = tid; idx < CI9 * 256; idx += 256) {
        int plane = idx >> 8;
        int pq = idx & 255, p = pq >> 4, q = pq & 15;
        int ci = plane / 9, r = plane - ci * 9, di = r / 3, dj = r - di * 3;
        int ii = i + di - 1, jj = j + dj - 1;
        if ((unsigned)ii < 16u && (unsigned)jj < 16u) {
            float v = x[(((size_t)b * CI + ci) << 16) + ((ii * 16 + jj) << 8) + pq];
            smem[plane * 324 + (p + 1) * 18 + (q + 1)] = f2bf(v);
        }
    }
    // ---- weights as B matrix [n][KP], bf16, zero padded ----
    for (int idx = tid; idx < 16 * KP; idx += 256) {
        int n = idx / KP, kk = idx - n * KP;
        unsigned short v = 0;
        if (n < CO && kk < KC) v = f2bf(w[n * KC + kk]);
        smem[wboff + idx] = v;
    }
    // ---- K -> LDS-offset table (pad K points at zero plane) ----
    for (int kk = tid; kk < KP; kk += 256) {
        int off;
        if (kk < KC) {
            int ci = kk / 81, t = kk - ci * 81;
            int di = t / 27, r1 = t - di * 27;
            int dj = r1 / 9, r2 = r1 - dj * 9;
            int dp = r2 / 3, dq = r2 - dp * 3;
            off = (ci * 9 + di * 3 + dj) * 324 + dp * 18 + dq;
        } else {
            off = CI9 * 324;                       // zero plane
        }
        smem[otoff + kk] = (unsigned short)off;
    }
    __syncthreads();

    const int lane = tid & 31, half = lane >> 4, lrow = lane & 15;
    const int wv = tid >> 5;
    const int m0 = wv * 32 + lrow;                 // A rows, tile0
    const int m1 = m0 + 16;                        // A rows, tile1
    const int pq0 = (m0 >> 4) * 18 + (m0 & 15);    // top-left of 3x3 window
    const int pq1 = (m1 >> 4) * 18 + (m1 & 15);

    v8f acc0 = (v8f){0.f, 0.f, 0.f, 0.f, 0.f, 0.f, 0.f, 0.f};
    v8f acc1 = (v8f){0.f, 0.f, 0.f, 0.f, 0.f, 0.f, 0.f, 0.f};

    for (int kb = 0; kb < KP; kb += 32) {
        // 16 K-offsets for this lane's half: {kb+h*8..+7} U {kb+16+h*8..+7}
        union { uint4 u; unsigned short s[8]; } o0, o1;
        o0.u = *reinterpret_cast<const uint4*>(&smem[otoff + kb + half * 8]);
        o1.u = *reinterpret_cast<const uint4*>(&smem[otoff + kb + 16 + half * 8]);

        union { unsigned short s[16]; v16bf v; } fa0, fa1;
        #pragma unroll
        for (int e = 0; e < 8; ++e) {
            int oa = o0.s[e], ob = o1.s[e];
            fa0.s[e]     = smem[pq0 + oa];
            fa0.s[8 + e] = smem[pq0 + ob];
            fa1.s[e]     = smem[pq1 + oa];
            fa1.s[8 + e] = smem[pq1 + ob];
        }
        union { uint4 u[2]; v16bf v; } fb;       // column n = lrow
        const unsigned short* wp = &smem[wboff + lrow * KP + kb + half * 8];
        fb.u[0] = *reinterpret_cast<const uint4*>(wp);
        fb.u[1] = *reinterpret_cast<const uint4*>(wp + 16);

        acc0 = __builtin_amdgcn_wmma_f32_16x16x32_bf16(
            false, fa0.v, false, fb.v, (short)0, acc0, false, false);
        acc1 = __builtin_amdgcn_wmma_f32_16x16x32_bf16(
            false, fa1.v, false, fb.v, (short)0, acc1, false, false);
    }

    // D layout: elem r: lanes0-15 -> M=r, lanes16-31 -> M=r+8; N = lane&15
    if (lrow < CO) {
        float bv = bias[lrow];
        float* yb = y + (((size_t)b * CO + lrow) << 16) + (ij << 8);
        #pragma unroll
        for (int r = 0; r < 8; ++r) {
            int ma = wv * 32 + half * 8 + r;
            yb[ma]      = fmaxf(acc0[r] + bv, 0.f);
            yb[ma + 16] = fmaxf(acc1[r] + bv, 0.f);
        }
    }
}

// transpose (b,i,j,p,q) -> (b,p,q,i,j), single channel
__global__ void transpose4_kernel(const float* __restrict__ x,
                                  float* __restrict__ xt) {
    int idx = blockIdx.x * 256 + threadIdx.x;
    int b = idx >> 16, s = (idx >> 8) & 255, t = idx & 255;
    xt[((size_t)b << 16) + (t << 8) + s] = x[idx];
}

// z = ya + transpose4(yb)
__global__ void combine_kernel(const float* __restrict__ ya,
                               const float* __restrict__ yb,
                               float* __restrict__ z) {
    int idx = blockIdx.x * 256 + threadIdx.x;
    int b = idx >> 16, s = (idx >> 8) & 255, t = idx & 255;
    z[idx] = ya[idx] + yb[((size_t)b << 16) + (t << 8) + s];
}

static inline size_t conv_shbytes(int CI) {
    int KC = CI * 81;
    int KP = ((KC + 31) >> 5) << 5;
    int planesz = (CI * 9 + 1) * 324;
    int wboff = (planesz + 7) & ~7;
    return (size_t)(wboff + 16 * KP + KP) * 2u;
}

// ---------------------------------------------------------------------------
extern "C" void kernel_launch(void* const* d_in, const int* in_sizes, int n_in,
                              void* d_out, int out_size, void* d_ws, size_t ws_size,
                              hipStream_t stream) {
    const float* fA = (const float*)d_in[0];
    const float* fB = (const float*)d_in[1];
    const float* w1 = (const float*)d_in[2];
    const float* b1 = (const float*)d_in[3];
    const float* w2 = (const float*)d_in[4];
    const float* b2 = (const float*)d_in[5];
    const float* w3 = (const float*)d_in[6];
    const float* b3 = (const float*)d_in[7];

    char* ws = (char*)d_ws;
    const size_t MB = 1ull << 20;
    unsigned short* AT  = (unsigned short*)(ws + 0);        // 16 MB
    unsigned short* BT  = (unsigned short*)(ws + 16 * MB);  // 16 MB
    float* corr   = (float*)(ws + 32 * MB);                 // 32 MB
    float* invA   = (float*)(ws + 64 * MB);                 // 32 KB
    float* invB   = (float*)(ws + 64 * MB + 32 * 1024);
    float* pooled = (float*)(ws + 64 * MB + 64 * 1024);     // 2 MB
    float* maxA   = (float*)(ws + 66 * MB + 64 * 1024);     // 8 KB
    float* maxB   = (float*)(ws + 66 * MB + 72 * 1024);     // 8 KB
    float* mmx    = (float*)(ws + 66 * MB + 80 * 1024);     // 2 MB
    float* xt     = (float*)(ws + 68 * MB + 80 * 1024);     // 2 MB
    float* y3a    = (float*)(ws + 70 * MB + 80 * 1024);     // 2 MB
    float* y3b    = (float*)(ws + 72 * MB + 80 * 1024);     // 2 MB
    float* zsum   = (float*)(ws + 74 * MB + 80 * 1024);     // 2 MB
    float* maxA2  = (float*)(ws + 76 * MB + 80 * 1024);     // 8 KB
    float* maxB2  = (float*)(ws + 76 * MB + 88 * 1024);     // 8 KB
    // reuse of dead regions for conv intermediates (20 MB each):
    float* y1 = (float*)(ws + 0);        // AT/BT dead after GEMM
    float* y2 = (float*)(ws + 32 * MB);  // corr dead after pooling/mm

    size_t sh1  = conv_shbytes(1);       // ~9.7 KB
    size_t sh10 = conv_shbytes(10);      // ~87 KB (fits gfx1250 320KB WGP LDS)

    // 1) norms
    invnorm_kernel<<<dim3(4, 8), 256, 0, stream>>>(fA, invA);
    invnorm_kernel<<<dim3(4, 8), 256, 0, stream>>>(fB, invB);
    // 2) normalize + transpose + bf16
    norm_transpose_kernel<<<dim3(32, 32, 8), dim3(32, 8), 0, stream>>>(fA, invA, AT);
    norm_transpose_kernel<<<dim3(32, 32, 8), dim3(32, 8), 0, stream>>>(fB, invB, BT);
    // 3) correlation GEMM (WMMA bf16)
    corr_gemm_kernel<<<dim3(8, 8, 8), 128, 0, stream>>>(AT, BT, corr);
    // 4) maxpool 2^4
    maxpool4d_kernel<<<2048, 256, 0, stream>>>(corr, pooled);
    // 5) mutual matching
    reduce_last_kernel<<<2048, 256, 0, stream>>>(pooled, maxA);
    reduce_mid_kernel<<<8, 256, 0, stream>>>(pooled, maxB);
    mm_apply_kernel<<<2048, 256, 0, stream>>>(pooled, maxA, maxB, mmx);
    // 6) neighbourhood consensus net, branch a: net(x)
    conv4d_wmma_kernel<<<dim3(256, 1, 8), 256, sh1,  stream>>>(mmx, w1, b1, y1, 1, 10);
    conv4d_wmma_kernel<<<dim3(256, 1, 8), 256, sh10, stream>>>(y1, w2, b2, y2, 10, 10);
    conv4d_wmma_kernel<<<dim3(256, 1, 8), 256, sh10, stream>>>(y2, w3, b3, y3a, 10, 1);
    // branch b: net(x^T)
    transpose4_kernel<<<2048, 256, 0, stream>>>(mmx, xt);
    conv4d_wmma_kernel<<<dim3(256, 1, 8), 256, sh1,  stream>>>(xt, w1, b1, y1, 1, 10);
    conv4d_wmma_kernel<<<dim3(256, 1, 8), 256, sh10, stream>>>(y1, w2, b2, y2, 10, 10);
    conv4d_wmma_kernel<<<dim3(256, 1, 8), 256, sh10, stream>>>(y2, w3, b3, y3b, 10, 1);
    combine_kernel<<<2048, 256, 0, stream>>>(y3a, y3b, zsum);
    // 7) final mutual matching -> output
    reduce_last_kernel<<<2048, 256, 0, stream>>>(zsum, maxA2);
    reduce_mid_kernel<<<8, 256, 0, stream>>>(zsum, maxB2);
    mm_apply_kernel<<<2048, 256, 0, stream>>>(zsum, maxA2, maxB2, (float*)d_out);
}
